// AdaptiveImageTokenPruner_7730941132810
// MI455X (gfx1250) — matrix operand order repaired and checked by reading, not compiled, not run on hardware
//
#include <hip/hip_runtime.h>

typedef __attribute__((ext_vector_type(16))) _Float16 v16h;
typedef __attribute__((ext_vector_type(8)))  _Float16 v8h;
typedef __attribute__((ext_vector_type(8)))  float    v8f;

#define H_DIM   1024   // hidden size of x / W1 inner dim
#define K_OUT   512    // W1 rows (scorer hidden)
#define N_TOK   576    // tokens per sample
#define SEGS    32     // 2^5 segments
#define SEG_LEN 18     // 576 / 32
#define MT      32     // tokens per workgroup in the score GEMM
#define LDS_STRIDE 520 // 512 + 8 halfs pad (keeps 16B alignment, staggers banks)

// ---------------------------------------------------------------------------
// Kernel 1: convert W1 (f32 [512,1024]) into f16, pre-swizzled into the exact
// per-lane fragment layout of V_WMMA_F32_16X16X32_F16's B operand (32x16, KxN):
//   element e of lane L holds B[K = (e&7) + 8*(L>=16) + 16*(e>=8)][N = L&15]
// Stored as [cbg 0..31][ks 0..31][chunk p 0..1][lane 0..31][8 halfs] so the
// GEMM loads each fragment with two fully coalesced 16B loads.
// ---------------------------------------------------------------------------
__global__ __launch_bounds__(256)
void w1_convert_kernel(const float* __restrict__ w1, _Float16* __restrict__ w1h) {
  int idx  = blockIdx.x * 256 + threadIdx.x;      // 0 .. 524287
  int j    =  idx        & 7;
  int lane = (idx >> 3)  & 31;
  int p    = (idx >> 8)  & 1;
  int ks   = (idx >> 9)  & 31;                    // K step of 32
  int cbg  = (idx >> 14) & 31;                    // column block of 16
  int kcol = cbg * 16 + (lane & 15);              // output column k' (N)
  int hdim = ks * 32 + j + 16 * p + 8 * (lane >> 4); // inner dim (K)
  w1h[idx] = (_Float16)w1[kcol * H_DIM + hdim];
}

// ---------------------------------------------------------------------------
// Kernel 2: fused scorer.  score[tok] = sum_k gelu(x.W1_k + b1_k) * W2_k
// (sigmoid / b2 / min-max normalization are monotonic -> dropped).
// WG = 256 threads = 8 waves; 32 tokens per WG; wave w owns columns
// [64w, 64w+64).  A panel staged in LDS as f16, K processed in 2 halves.
// ---------------------------------------------------------------------------
__global__ __launch_bounds__(256)
void score_kernel(const float* __restrict__ x,        // [B*N, 1024]
                  const _Float16* __restrict__ w1h,   // swizzled f16 W1
                  const float* __restrict__ b1,       // [512]
                  const float* __restrict__ w2,       // [512]
                  float* __restrict__ scores)         // [B*N]
{
  __shared__ __align__(16) _Float16 lA[MT * LDS_STRIDE];
  __shared__ float lPart[8 * MT];

  const int tid     = threadIdx.x;
  const int wave    = tid >> 5;
  const int lane    = tid & 31;
  const int tokbase = blockIdx.x * MT;

  v8f acc[4][2];
#pragma unroll
  for (int cb = 0; cb < 4; ++cb)
#pragma unroll
    for (int mt = 0; mt < 2; ++mt)
      acc[cb][mt] = v8f{0.f, 0.f, 0.f, 0.f, 0.f, 0.f, 0.f, 0.f};

  for (int kh = 0; kh < 2; ++kh) {
    // --- cooperative load: 32 rows x 512 cols f32 -> f16 LDS (coalesced) ---
    {
      const int row  = tid >> 3;            // 0..31
      const int col0 = (tid & 7) * 64;      // 0..448
      const float4* src4 =
          (const float4*)(x + (size_t)(tokbase + row) * H_DIM + kh * 512 + col0);
      _Float16* dst = lA + row * LDS_STRIDE + col0;
#pragma unroll
      for (int i = 0; i < 8; ++i) {
        float4 a = src4[2 * i];
        float4 b = src4[2 * i + 1];
        v8h h = {(_Float16)a.x, (_Float16)a.y, (_Float16)a.z, (_Float16)a.w,
                 (_Float16)b.x, (_Float16)b.y, (_Float16)b.z, (_Float16)b.w};
        *(v8h*)(dst + i * 8) = h;           // ds_store_b128
      }
    }
    __syncthreads();

    // --- WMMA main loop: 4 column blocks x 16 K-steps x 2 M-tiles ---
#pragma unroll
    for (int cb = 0; cb < 4; ++cb) {
      const int cbg = wave * 4 + cb;
      for (int ks = 0; ks < 16; ++ks) {
        const int ksg = kh * 16 + ks;
        // B fragment: two coalesced 16B global loads from swizzled W1
        const v8h* bp = (const v8h*)w1h + ((size_t)(cbg * 32 + ksg) * 2) * 32 + lane;
        v16h bfrag;
        ((v8h*)&bfrag)[0] = bp[0];
        ((v8h*)&bfrag)[1] = bp[32];
#pragma unroll
        for (int mt = 0; mt < 2; ++mt) {
          // A fragment: lane L -> row L&15, K base = 32*ks + 8*(L>=16),
          // elements 0..7 = K base..base+7, elements 8..15 = K base+16..base+23
          const _Float16* ap = lA + (mt * 16 + (lane & 15)) * LDS_STRIDE
                                  + ks * 32 + ((lane >> 4) << 3);
          v16h afrag;
          ((v8h*)&afrag)[0] = *(const v8h*)(ap);        // ds_load_b128
          ((v8h*)&afrag)[1] = *(const v8h*)(ap + 16);   // ds_load_b128
          acc[cb][mt] = __builtin_amdgcn_wmma_f32_16x16x32_f16(
              false, afrag, false, bfrag, (short)0, acc[cb][mt], false, false);
        }
      }
    }
    __syncthreads();   // A panel dead before next half is loaded
  }

  // --- epilogue: bias + exact GELU + W2 weighting, fused into score ---
  float b1v[4], w2v[4];
#pragma unroll
  for (int cb = 0; cb < 4; ++cb) {
    int kcol = wave * 64 + cb * 16 + (lane & 15);
    b1v[cb] = b1[kcol];
    w2v[cb] = w2[kcol];
  }

  float sacc[2][8];
#pragma unroll
  for (int mt = 0; mt < 2; ++mt)
#pragma unroll
    for (int v = 0; v < 8; ++v) sacc[mt][v] = 0.f;

#pragma unroll
  for (int cb = 0; cb < 4; ++cb)
#pragma unroll
    for (int mt = 0; mt < 2; ++mt)
#pragma unroll
      for (int v = 0; v < 8; ++v) {
        float t = acc[cb][mt][v] + b1v[cb];
        float g = 0.5f * t * (1.f + erff(t * 0.70710678118654752f)); // exact GELU
        sacc[mt][v] += g * w2v[cb];
      }

  // reduce over the 16 columns (lanes differing in bits 0..3)
#pragma unroll
  for (int off = 1; off < 16; off <<= 1)
#pragma unroll
    for (int mt = 0; mt < 2; ++mt)
#pragma unroll
      for (int v = 0; v < 8; ++v)
        sacc[mt][v] += __shfl_xor(sacc[mt][v], off, 32);

  // C layout: VGPR v, lane L -> row M = v + 8*(L>=16); write wave partials
  if ((lane & 15) == 0) {
    int rbase = (lane >> 4) * 8;
#pragma unroll
    for (int mt = 0; mt < 2; ++mt)
#pragma unroll
      for (int v = 0; v < 8; ++v)
        lPart[wave * MT + mt * 16 + rbase + v] = sacc[mt][v];
  }
  __syncthreads();

  if (tid < MT) {
    float s = 0.f;
#pragma unroll
    for (int w = 0; w < 8; ++w) s += lPart[w * MT + tid];
    scores[tokbase + tid] = s;   // deterministic (no atomics)
  }
}

// ---------------------------------------------------------------------------
// Kernel 3: per-segment top-quota selection.  One thread per (batch, segment).
// Strict '>' scan in ascending index order == jax.lax.top_k tie-breaking.
// Emitting chosen indices in ascending i gives the globally sorted order.
// ---------------------------------------------------------------------------
__global__ __launch_bounds__(256)
void topk_kernel(const float* __restrict__ scores, int* __restrict__ sel,
                 int quota, int nB) {
  int sid = blockIdx.x * blockDim.x + threadIdx.x;
  if (sid >= nB * SEGS) return;
  int b = sid >> 5, seg = sid & 31;
  const float* s = scores + (size_t)b * N_TOK + seg * SEG_LEN;
  float v[SEG_LEN];
  bool  used[SEG_LEN];
#pragma unroll
  for (int i = 0; i < SEG_LEN; ++i) { v[i] = s[i]; used[i] = false; }
  for (int q = 0; q < quota; ++q) {
    float best = -3.402823466e+38f;
    int   bi   = 0;
    for (int i = 0; i < SEG_LEN; ++i)
      if (!used[i] && v[i] > best) { best = v[i]; bi = i; }
    used[bi] = true;
  }
  int* o = sel + (size_t)b * (SEGS * quota) + seg * quota;
  int c = 0;
  for (int i = 0; i < SEG_LEN; ++i)
    if (used[i]) o[c++] = seg * SEG_LEN + i;
}

// ---------------------------------------------------------------------------
// Kernel 4: gather selected rows.  One block per output row, float4 streaming.
// ---------------------------------------------------------------------------
__global__ __launch_bounds__(256)
void gather_kernel(const float* __restrict__ x, const int* __restrict__ sel,
                   float* __restrict__ out, int T) {
  int row = blockIdx.x;            // b*T + t
  int b   = row / T;
  int src = sel[row];
  const float4* xp = (const float4*)(x + ((size_t)b * N_TOK + src) * H_DIM);
  float4*       op = (float4*)(out + (size_t)row * H_DIM);
  op[threadIdx.x] = xp[threadIdx.x];   // 256 threads x 16B = 4KB row
}

// ---------------------------------------------------------------------------
extern "C" void kernel_launch(void* const* d_in, const int* in_sizes, int n_in,
                              void* d_out, int out_size, void* d_ws, size_t ws_size,
                              hipStream_t stream) {
  const float* x  = (const float*)d_in[0];   // [B, 576, 1024] f32
  const float* W1 = (const float*)d_in[1];   // [512, 1024]
  const float* b1 = (const float*)d_in[2];   // [512]
  const float* W2 = (const float*)d_in[3];   // [1, 512]
  // d_in[4] (b2) and d_in[5] (target) unused: b2 is a monotonic shift; the
  // selection count is derived from out_size (avoids device-scalar reads
  // during graph capture).

  const int B = in_sizes[0] / (N_TOK * H_DIM);              // 256
  const int T = (int)((long long)out_size / H_DIM / B);     // 288
  const int quota = T / SEGS;                               // 9

  _Float16* w1h   = (_Float16*)d_ws;                                   // 1 MB
  float*    scores = (float*)((char*)d_ws + (size_t)K_OUT * H_DIM * 2); // B*N f32
  int*      sel    = (int*)((char*)scores + (size_t)B * N_TOK * sizeof(float));

  w1_convert_kernel<<<(K_OUT * H_DIM) / 256, 256, 0, stream>>>(W1, w1h);
  score_kernel<<<(B * N_TOK) / MT, 256, 0, stream>>>(x, w1h, b1, W2, scores);
  topk_kernel<<<(B * SEGS + 255) / 256, 256, 0, stream>>>(scores, sel, quota, B);
  gather_kernel<<<B * T, 256, 0, stream>>>(x, sel, (float*)d_out, T);
}
